// LSTM_69672959476112
// MI455X (gfx1250) — compile-verified, matrix operand order checked by reading, not run on hardware
//
#include <hip/hip_runtime.h>
#include <hip/hip_bf16.h>

#define BB 1024
#define TT 256
#define HH 256
#define CC 10

typedef __attribute__((ext_vector_type(8))) int   v8i;
typedef __attribute__((ext_vector_type(8))) float v8f;

// ---------------------------------------------------------------------------
// Branch-free f32 -> OCP FP8 E4M3 (bias 7, max 448), round-to-nearest-even.
// Normal path: RNE on the f32 bit pattern (add half-ulp-at-bit-20 + lsb),
// then rebias: e4m3 bits = (u>>20) - (120<<3).  Subnormal path (<2^-6):
// rintf(a*512) in 0..8 (8 aliases the encoding of 2^-6).  One select.
// ---------------------------------------------------------------------------
__device__ __forceinline__ unsigned char f32_to_e4m3(float f) {
    unsigned sgn = (__float_as_uint(f) >> 24) & 0x80u;
    float a = fminf(fabsf(f), 448.0f);
    unsigned u  = __float_as_uint(a);
    unsigned un = u + 0x7FFFFu + ((u >> 20) & 1u);     // RNE round at bit 20
    unsigned nrm = (un >> 20) - 960u;                  // ((exp-120)<<3)|mant3
    unsigned sub = (unsigned)(int)rintf(a * 512.0f);   // 0..8, quantum 2^-9
    unsigned mag = (a >= 0.015625f) ? nrm : sub;
    return (unsigned char)(sgn | mag);
}

// Branch-free activations on v_exp_f32 (saturate correctly via exp->0/inf).
__device__ __forceinline__ float fast_sigmoid(float x) {
    return __fdividef(1.0f, 1.0f + __expf(-x));
}
__device__ __forceinline__ float fast_tanh(float x) {
    return 1.0f - __fdividef(2.0f, 1.0f + __expf(2.0f * x));
}

// ---------------------------------------------------------------------------
// Prologue: the four recurrent matrices (f32 row-major 256x256,
// out[n] = sum_k h[k]*W[k][n]) -> E4M3 B-fragments for
// v_wmma_f32_16x16x64_fp8_fp8.  8-bit fragment layout (ISA 7.12.2, N across
// lanes):  byte index = (((gate*16+n)*4 + kt)*32 + lane)*32 + v*4 + b
//          K = kt*64 + (v&1)*4 + ((v>>1)&1)*16 + (v>>2)*32 + 8*(lane>>4) + b
//          N = n*16 + (lane&15)
// One thread per byte: 4*16*4*32*32 = 262144.
// ---------------------------------------------------------------------------
__global__ void swizzleB8(const float* __restrict__ Wg, const float* __restrict__ Wi,
                          const float* __restrict__ Wf, const float* __restrict__ Wo,
                          unsigned char* __restrict__ frag) {
    int idx  = blockIdx.x * 256 + threadIdx.x;     // 0 .. 262143
    int b    =  idx        & 3;
    int v    = (idx >> 2)  & 7;
    int lane = (idx >> 5)  & 31;
    int kt   = (idx >> 10) & 3;
    int n    = (idx >> 12) & 15;
    int gate = (idx >> 16) & 3;
    int lh = lane >> 4, ll = lane & 15;
    int k   = kt * 64 + (v & 1) * 4 + ((v >> 1) & 1) * 16 + (v >> 2) * 32 + 8 * lh + b;
    int col = n * 16 + ll;
    const float* W = (gate == 0) ? Wg : (gate == 1) ? Wi : (gate == 2) ? Wf : Wo;
    frag[idx] = f32_to_e4m3(W[k * HH + col]);
}

// ---------------------------------------------------------------------------
// Persistent LSTM: block = 256 threads (8 waves), owns 16 batch rows for all
// 256 timesteps.  All weight fragments live in registers (32 x v8i = 256
// VGPRs/wave) for the entire recurrence; h lives in LDS as E4M3 bytes; c in
// registers; the block's x tile is staged to LDS once.  Wave w computes
// column tiles {2w, 2w+1} for all 4 gates -> cell update is wave-local.
// ---------------------------------------------------------------------------
__global__ void __launch_bounds__(256) lstm_main(
    const float* __restrict__ x, const v8i* __restrict__ fragB,
    const float* __restrict__ Wgx, const float* __restrict__ Wix,
    const float* __restrict__ Wfx, const float* __restrict__ Wox,
    const float* __restrict__ bg,  const float* __restrict__ bi,
    const float* __restrict__ bf,  const float* __restrict__ bo,
    const float* __restrict__ Wph, const float* __restrict__ bp,
    float* __restrict__ out)
{
    __shared__ unsigned char hbuf[16 * HH];   // 4 KB : h_t tile in E4M3 [16][256]
    __shared__ float         hfl[16 * TT];    // 16 KB: x tile [16][256] during the
                                              // loop, final f32 h afterwards
    const int tid  = threadIdx.x;
    const int lane = tid & 31;
    const int wv   = tid >> 5;           // wave id 0..7
    const int ll   = lane & 15;
    const int lh   = lane >> 4;
    const int row0 = blockIdx.x * 16;    // global batch row base

    // Resident weight fragments: [slot s][gate][kt]  (loaded once, stay in VGPRs)
    v8i Bres[2][4][4];
    #pragma unroll
    for (int s = 0; s < 2; ++s)
        #pragma unroll
        for (int g = 0; g < 4; ++g)
            #pragma unroll
            for (int kt = 0; kt < 4; ++kt)
                Bres[s][g][kt] = fragB[((g * 16 + (2 * wv + s)) * 4 + kt) * 32 + lane];

    // Per-thread x-weights and biases for the two column slots this lane owns.
    float wxv[2][4], bv[2][4];
    #pragma unroll
    for (int s = 0; s < 2; ++s) {
        int col = (2 * wv + s) * 16 + ll;
        wxv[s][0] = Wgx[col]; wxv[s][1] = Wix[col];
        wxv[s][2] = Wfx[col]; wxv[s][3] = Wox[col];
        bv[s][0]  = bg[col];  bv[s][1]  = bi[col];
        bv[s][2]  = bf[col];  bv[s][3]  = bo[col];
    }

    // Stage this block's x tile (16 rows x 256 steps, contiguous) into LDS.
    {
        const float* xg = x + (size_t)row0 * TT;   // [16][256] row-major
        #pragma unroll
        for (int i = 0; i < 16; ++i)
            hfl[i * 256 + tid] = xg[i * 256 + tid];
    }

    float creg[2][8], hfin[2][8];
    #pragma unroll
    for (int s = 0; s < 2; ++s)
        #pragma unroll
        for (int r = 0; r < 8; ++r) { creg[s][r] = 0.0f; hfin[s][r] = 0.0f; }

    // h_0 = 0
    #pragma unroll
    for (int j = 0; j < 16; ++j) hbuf[tid * 16 + j] = 0;

    for (int t = 0; t < TT; ++t) {
        __syncthreads();   // hbuf (h_t) ready; x tile ready on first iteration

        // x values for this lane's 8 rows (LDS broadcast reads).
        float xrow[8];
        #pragma unroll
        for (int r = 0; r < 8; ++r) xrow[r] = hfl[(8 * lh + r) * 256 + t];

        // A fragments: 16x64 E4M3 tiles of h from LDS (ISA 8-bit A layout).
        // Lane l row M=l&15; v0,v1 <- K 8lh+0..7 ; v2,v3 <- +16 ; v4..7 <- +32.
        v8i aF[4];
        #pragma unroll
        for (int kt = 0; kt < 4; ++kt) {
            const unsigned char* rp = &hbuf[ll * HH + kt * 64 + 8 * lh];
            uint2 c0 = *(const uint2*)(rp);        // ds_load_b64 (pairs merge)
            uint2 c1 = *(const uint2*)(rp + 16);
            uint2 c2 = *(const uint2*)(rp + 32);
            uint2 c3 = *(const uint2*)(rp + 48);
            v8i a;
            a[0] = (int)c0.x; a[1] = (int)c0.y;
            a[2] = (int)c1.x; a[3] = (int)c1.y;
            a[4] = (int)c2.x; a[5] = (int)c2.y;
            a[6] = (int)c3.x; a[7] = (int)c3.y;
            aF[kt] = a;
        }

        // Gate pre-activations: acc[s][g] = (h_t @ W_g)[rows, coltile 2w+s]
        v8f acc[2][4];
        #pragma unroll
        for (int s = 0; s < 2; ++s)
            #pragma unroll
            for (int g = 0; g < 4; ++g)
                acc[s][g] = (v8f){0.f,0.f,0.f,0.f,0.f,0.f,0.f,0.f};

        #pragma unroll
        for (int s = 0; s < 2; ++s)
            #pragma unroll
            for (int g = 0; g < 4; ++g)
                #pragma unroll
                for (int kt = 0; kt < 4; ++kt)
                    acc[s][g] = __builtin_amdgcn_wmma_f32_16x16x64_fp8_fp8(
                        aF[kt], Bres[s][g][kt], (short)0, acc[s][g], false, false);

        __syncthreads();   // all hbuf (and t=255 x) reads done; safe to overwrite

        // Elementwise cell update (wave-local, branch-free).  D layout:
        // VGPR r, lane l -> row M = r + 8*(l>>4), col N = l&15 of tile 2w+s.
        #pragma unroll
        for (int s = 0; s < 2; ++s) {
            #pragma unroll
            for (int r = 0; r < 8; ++r) {
                const int   rowl = 8 * lh + r;
                const float xv   = xrow[r];
                float pg = acc[s][0][r] + xv * wxv[s][0] + bv[s][0];
                float pi = acc[s][1][r] + xv * wxv[s][1] + bv[s][1];
                float pf = acc[s][2][r] + xv * wxv[s][2] + bv[s][2];
                float po = acc[s][3][r] + xv * wxv[s][3] + bv[s][3];
                float g_ = fast_tanh(pg);
                float i_ = fast_sigmoid(pi);
                float f_ = fast_sigmoid(pf);
                float o_ = fast_sigmoid(po);
                float cn = g_ * i_ + creg[s][r] * f_;
                creg[s][r] = cn;
                float hn = fast_tanh(cn) * o_;
                hfin[s][r] = hn;
                hbuf[rowl * HH + (2 * wv + s) * 16 + ll] = f32_to_e4m3(hn);
            }
        }
    }

    __syncthreads();   // everyone past the loop before hfl is repurposed
    // Dump final h (f32) for the projection (fully overwrites the x stage).
    #pragma unroll
    for (int s = 0; s < 2; ++s)
        #pragma unroll
        for (int r = 0; r < 8; ++r)
            hfl[(8 * lh + r) * HH + (2 * wv + s) * 16 + ll] = hfin[s][r];
    __syncthreads();

    // Projection: out[16 rows][10 cols] = h_last @ Wph + bp
    if (tid < 16 * CC) {
        int r = tid / CC, cc = tid % CC;
        float sum = bp[cc];
        #pragma unroll 8
        for (int k = 0; k < HH; ++k)
            sum += hfl[r * HH + k] * Wph[k * CC + cc];
        out[(row0 + r) * CC + cc] = sum;
    }
}

// ---------------------------------------------------------------------------
extern "C" void kernel_launch(void* const* d_in, const int* in_sizes, int n_in,
                              void* d_out, int out_size, void* d_ws, size_t ws_size,
                              hipStream_t stream) {
    const float* x   = (const float*)d_in[0];
    const float* Wgx = (const float*)d_in[1];
    const float* Wgh = (const float*)d_in[2];
    const float* Wix = (const float*)d_in[3];
    const float* Wih = (const float*)d_in[4];
    const float* Wfx = (const float*)d_in[5];
    const float* Wfh = (const float*)d_in[6];
    const float* Wox = (const float*)d_in[7];
    const float* Woh = (const float*)d_in[8];
    const float* Wph = (const float*)d_in[9];
    const float* bg  = (const float*)d_in[10];
    const float* bi  = (const float*)d_in[11];
    const float* bf  = (const float*)d_in[12];
    const float* bo  = (const float*)d_in[13];
    const float* bp  = (const float*)d_in[14];

    unsigned char* frag = (unsigned char*)d_ws;   // 256 KB of E4M3 B-fragments

    // 262144 fragment bytes, one thread each.
    swizzleB8<<<1024, 256, 0, stream>>>(Wgh, Wih, Wfh, Woh, frag);

    lstm_main<<<BB / 16, 256, 0, stream>>>(
        x, (const v8i*)d_ws,
        Wgx, Wix, Wfx, Wox, bg, bi, bf, bo, Wph, bp, (float*)d_out);
}
